// LSTM_79826262164154
// MI455X (gfx1250) — matrix operand order repaired and checked by reading, not compiled, not run on hardware
//
#include <hip/hip_runtime.h>
#include <hip/hip_bf16.h>
#include <stdint.h>

typedef __bf16 bf16;
typedef __attribute__((ext_vector_type(16))) __bf16 v16bf;
typedef __attribute__((ext_vector_type(8)))  __bf16 v8bf;
typedef __attribute__((ext_vector_type(8)))  float  v8f;
typedef __attribute__((ext_vector_type(4)))  unsigned int u32x4;
typedef __attribute__((ext_vector_type(8)))  int i32x8;
typedef __attribute__((ext_vector_type(4)))  int i32x4;

#define TSTEPS 512
#define BATCH  256
#define HID    1024
#define INW    64
#define NWG_TOTAL 256
#define KBLK   256          // K-block staged per TDM transfer (64 rows x 256 cols bf16 = 32KB)

#if defined(__has_builtin)
#  if __has_builtin(__builtin_amdgcn_tensor_load_to_lds) && __has_builtin(__builtin_amdgcn_s_wait_tensorcnt)
#    define TDM_OK 1
#  endif
#endif
#ifndef TDM_OK
#  define TDM_OK 0
#endif

// ---------------------------------------------------------------------------
#if TDM_OK
// Issue a TDM 2D tile load: global (row-major, row stride in elems) -> packed LDS.
// D# per CDNA5 ISA ch.8: group0 = flags/lds_addr/global_addr/type, group1 = dims.
__device__ __forceinline__ void tdm_load_2d(unsigned lds_byte_off,
                                            const bf16* gsrc,
                                            unsigned width_elems,   // tile_dim0
                                            unsigned rows,          // tile_dim1
                                            unsigned row_stride_elems)
{
  uint64_t ga = (uint64_t)(uintptr_t)gsrc;
  union { unsigned u[4]; u32x4 v; } g0;
  g0.u[0] = 1u;                                           // count=1 (valid), user mode
  g0.u[1] = lds_byte_off;                                 // lds_addr [63:32]
  g0.u[2] = (unsigned)(ga & 0xffffffffu);                 // global_addr[95:64]
  g0.u[3] = (unsigned)((ga >> 32) & 0x01ffffffu) | 0x80000000u; // addr[56:32] | type=2

  uint64_t q0 = 0, q1 = 0, q2 = 0, q3 = 0;
  q0 |= (uint64_t)1u << 16;                               // data_size = 1 -> 2 bytes
  q0 |= (uint64_t)width_elems << 48;                      // tensor_dim0 [79:48] (low)
  q1 |= (uint64_t)width_elems >> 16;                      //              (high)
  q1 |= (uint64_t)rows << 16;                             // tensor_dim1 [111:80]
  q1 |= (uint64_t)(width_elems & 0xffffu) << 48;          // tile_dim0 [127:112]
  q2 |= (uint64_t)(rows & 0xffffu);                       // tile_dim1 [143:128]
  q2 |= (uint64_t)row_stride_elems << 32;                 // tensor_dim0_stride [207:160] (low)
  q3 |= (uint64_t)row_stride_elems >> 32;                 //                     (high)
  union { uint64_t q[4]; i32x8 v; } g1;
  g1.q[0] = q0; g1.q[1] = q1; g1.q[2] = q2; g1.q[3] = q3;

  i32x4 z4 = (i32x4){0, 0, 0, 0};
#if __clang_major__ >= 23
  i32x8 z8 = (i32x8){0, 0, 0, 0, 0, 0, 0, 0};
  __builtin_amdgcn_tensor_load_to_lds(g0.v, g1.v, z4, z4, z8, 0);
#else
  __builtin_amdgcn_tensor_load_to_lds(g0.v, g1.v, z4, z4, 0);
#endif
}
#endif

// ---------------------------------------------------------------------------
// Fragment loaders (CDNA5 WMMA 16x16x32 bf16 layouts, ISA 7.12.2)
// A (from LDS panel, packed [rows][ld]): lane L: row = row0 + (L&15),
//   K = k0 + 8*(L>>4) + {0..7} and +16 more.
// Non-volatile: the guard store in the kernel marks lds_pool as written, so
// these lower to ds_load_b128 (AS3 inferred) and cannot be folded to poison.
__device__ __forceinline__ v16bf load_a_frag_lds(const bf16* lA, int ld,
                                                 int row0, int k0, int lane) {
  int m  = row0 + (lane & 15);
  int kb = k0 + ((lane >> 4) << 3);
  const bf16* p = lA + m * ld + kb;
  union { v16bf v; v8bf h[2]; } u;
  u.h[0] = *(const v8bf*)(p);        // ds_load_b128
  u.h[1] = *(const v8bf*)(p + 16);   // ds_load_b128
  return u.v;
}

// B: 32(K) x 16(N): lane L: col N = n0 + (L&15), K = k0 + 8*(L>>4) + {0..7} and +16 more
// W row-major [4H][K] -> two 16B global loads per lane.
__device__ __forceinline__ v16bf load_b_frag(const bf16* __restrict__ W, int ld,
                                             int n0, int k0, int lane) {
  int n  = n0 + (lane & 15);
  int kb = k0 + ((lane >> 4) << 3);
  const bf16* p = W + (size_t)n * ld + kb;
  union { v16bf v; v8bf h[2]; } u;
  u.h[0] = *(const v8bf*)(p);
  u.h[1] = *(const v8bf*)(p + 16);
  return u.v;
}

__device__ __forceinline__ float fast_sigmoid(float x) {
  return 1.0f / (1.0f + __expf(-x));
}

// ---------------------------------------------------------------------------
// GEMM over one activation panel (this WG's 64 batch rows x Kext cols),
// TDM-staged through double-buffered LDS, accumulating into acc[4] (i,f,g,o).
__device__ __forceinline__ void gemm_panel(
    const bf16* __restrict__ Asrc, int ldA, int Kext,  // global panel base (row 0 of WG tile)
    const bf16* __restrict__ W, int ldW, int hbase,    // weights [4H][ldW]
    v8f* acc, bf16* lds_pool, int mrow, int lane)
{
  const int nblk = (Kext + KBLK - 1) / KBLK;
#if TDM_OK
  if (threadIdx.x == 0) {
    int bw0 = Kext < KBLK ? Kext : KBLK;
    tdm_load_2d(0u, Asrc, (unsigned)bw0, 64u, (unsigned)ldA);
  }
#endif
  for (int blk = 0; blk < nblk; ++blk) {
    const int kb = blk * KBLK;
    const int bw = (Kext - kb) < KBLK ? (Kext - kb) : KBLK;
    bf16* lA = lds_pool + (blk & 1) * (64 * KBLK);
#if TDM_OK
    if (threadIdx.x == 0) {
      if (blk + 1 < nblk) {                      // prefetch next block, then wait for current
        int kn = kb + KBLK;
        int bwn = (Kext - kn) < KBLK ? (Kext - kn) : KBLK;
        tdm_load_2d((unsigned)((((blk + 1) & 1) * (64 * KBLK)) * 2),
                    Asrc + kn, (unsigned)bwn, 64u, (unsigned)ldA);
        __builtin_amdgcn_s_wait_tensorcnt(1);
      } else {
        __builtin_amdgcn_s_wait_tensorcnt(0);
      }
    }
    __syncthreads();
    asm volatile("" ::: "memory");   // TDM wrote LDS: order it before the reads below
#else
    __syncthreads();
    for (int i = (int)threadIdx.x * 8; i < 64 * bw; i += (int)blockDim.x * 8) {
      int r = i / bw, c = i % bw;
      *(v8bf*)(lA + r * bw + c) = *(const v8bf*)(Asrc + (size_t)r * ldA + kb + c);
    }
    __syncthreads();
#endif
    for (int k0 = 0; k0 < bw; k0 += 32) {
      v16bf a = load_a_frag_lds(lA, bw, mrow, k0, lane);
#pragma unroll
      for (int g = 0; g < 4; ++g) {
        v16bf b = load_b_frag(W, ldW, g * HID + hbase, kb + k0, lane);
        acc[g] = __builtin_amdgcn_wmma_f32_16x16x32_bf16(
            false, a, false, b, (short)0, acc[g], false, false);
      }
    }
    __syncthreads();   // all waves done reading lA before it is overwritten
  }
}

// ---------------------------------------------------------------------------
// One LSTM cell step for this wave's 16(batch) x 16(hidden) tile; c in registers.
__device__ __forceinline__ void lstm_tile_step(
    const bf16* __restrict__ AxP, int ldx, int KX,     // input panel (WG's 64 rows)
    const bf16* __restrict__ AhP,                      // recurrent panel (WG's 64 rows)
    const bf16* __restrict__ Wx, const bf16* __restrict__ Wh,
    const float* __restrict__ bias,
    bf16* __restrict__ h_out, float* __restrict__ h_out_f32,
    v8f& cc, bf16* lds_pool,
    int mbase, int mrow, int hbase, int lane)
{
  v8f acc[4] = {};   // i, f, g, o
  gemm_panel(AxP, ldx, KX,  Wx, KX,  hbase, acc, lds_pool, mrow, lane);
  gemm_panel(AhP, HID, HID, Wh, HID, hbase, acc, lds_pool, mrow, lane);

  // C/D layout: acc[r] -> M = mbase + r + 8*(lane>>4), N = hbase + (lane&15)
  int nl = lane & 15;
  int mo = (lane >> 4) * 8;
  float bi  = bias[0 * HID + hbase + nl];
  float bf_ = bias[1 * HID + hbase + nl];
  float bg  = bias[2 * HID + hbase + nl];
  float bo  = bias[3 * HID + hbase + nl];
#pragma unroll
  for (int r = 0; r < 8; ++r) {
    float iv = fast_sigmoid(acc[0][r] + bi);
    float fv = fast_sigmoid(acc[1][r] + bf_);
    float gv = tanhf(acc[2][r] + bg);
    float ov = fast_sigmoid(acc[3][r] + bo);
    float c  = fv * cc[r] + iv * gv;
    cc[r] = c;
    float h = ov * tanhf(c);
    size_t m = (size_t)(mbase + r + mo);
    h_out[m * HID + hbase + nl] = (bf16)h;
    if (h_out_f32) h_out_f32[m * HID + hbase + nl] = h;
  }
}

// ---------------------------------------------------------------------------
__device__ __forceinline__ void grid_barrier(unsigned* cnt, unsigned target) {
  __threadfence();
  __syncthreads();
  if (threadIdx.x == 0) {
    __hip_atomic_fetch_add(cnt, 1u, __ATOMIC_ACQ_REL, __HIP_MEMORY_SCOPE_AGENT);
    while (__hip_atomic_load(cnt, __ATOMIC_ACQUIRE, __HIP_MEMORY_SCOPE_AGENT) < target) {
      __builtin_amdgcn_s_sleep(2);
    }
  }
  __syncthreads();
}

// ---------------------------------------------------------------------------
// Persistent kernel: 256 WGs x 256 threads. WGs 0..127 = layer0, 128..255 = layer1.
// Phase p: layer0 runs step t=p, layer1 runs step t=p-1 (software pipeline).
__global__ __launch_bounds__(256, 1) void lstm_persistent(
    const bf16* __restrict__ x_bf,   // [T][B][I]
    const bf16* __restrict__ wx0, const bf16* __restrict__ wh0,
    const bf16* __restrict__ wx1, const bf16* __restrict__ wh1,
    const float* __restrict__ b0, const float* __restrict__ b1,
    bf16* __restrict__ h0_bf,        // [2][B][H] double buffer
    bf16* __restrict__ h1_bf,        // [2][B][H] double buffer
    float* __restrict__ h1_f32,      // [B][H] final layer-1 hidden
    unsigned* barrier_cnt,
    int guard)                       // always 0 (host): keeps LDS "written" for the compiler
{
  // 64KB LDS: double-buffered 64x256 bf16 activation panel, at LDS offset 0.
  __shared__ __attribute__((aligned(64))) bf16 lds_pool[2 * 64 * KBLK];
  if (guard) {                      // never taken; defeats never-written-LDS folding
    lds_pool[threadIdx.x] = (bf16)1.0f;
  }

  const int wg    = blockIdx.x;
  const int layer = wg >> 7;
  const int tid   = wg & 127;
  const int mtile = tid & 3;          // 4 x 64 batch rows
  const int htile = tid >> 2;         // 32 x 32 hidden cols
  const int wave  = (int)threadIdx.x >> 5;
  const int lane  = (int)threadIdx.x & 31;
  const int mrow  = (wave & 3) * 16;          // row inside this WG's 64-row panel
  const int mbase = mtile * 64 + mrow;        // absolute batch row
  const int hbase = htile * 32 + (wave >> 2) * 16;
  const unsigned nwg = gridDim.x;
  const size_t HBUF = (size_t)BATCH * HID;
  const size_t prow = (size_t)mtile * 64;     // panel start row

  v8f cc = {};   // register-resident cell state for this tile (all 512 steps)

  for (int p = 0; p <= TSTEPS; ++p) {
    if (layer == 0) {
      if (p < TSTEPS) {
        const bf16* AxP = x_bf + ((size_t)p * BATCH + prow) * INW;
        const bf16* AhP = h0_bf + ((p + 1) & 1) * HBUF + prow * HID;  // h0[p-1]
        bf16* hout      = h0_bf + (p & 1) * HBUF;                     // h0[p]
        lstm_tile_step(AxP, INW, INW, AhP, wx0, wh0, b0,
                       hout, nullptr, cc, lds_pool, mbase, mrow, hbase, lane);
      }
    } else {
      int t = p - 1;
      if (t >= 0) {
        const bf16* AxP = h0_bf + (t & 1) * HBUF + prow * HID;        // h0[t]
        const bf16* AhP = h1_bf + ((t + 1) & 1) * HBUF + prow * HID;  // h1[t-1]
        bf16* hout      = h1_bf + (t & 1) * HBUF;                     // h1[t]
        float* hf = (t == TSTEPS - 1) ? h1_f32 : nullptr;
        lstm_tile_step(AxP, HID, HID, AhP, wx1, wh1, b1,
                       hout, hf, cc, lds_pool, mbase, mrow, hbase, lane);
      }
    }
    grid_barrier(barrier_cnt, nwg * (unsigned)(p + 1));
  }
}

// ---------------------------------------------------------------------------
__global__ void init_kernel(unsigned* cnt, bf16* h0, bf16* h1, long n) {
  long i = (long)blockIdx.x * blockDim.x + threadIdx.x;
  if (i == 0) *cnt = 0u;
  if (i < n) { h0[i] = (bf16)0.0f; h1[i] = (bf16)0.0f; }
}

// x [B][T][I] fp32 -> x_bf [T][B][I] bf16
__global__ void convert_x_kernel(const float* __restrict__ x, bf16* __restrict__ x_bf) {
  long i = (long)blockIdx.x * blockDim.x + threadIdx.x;
  if (i < (long)BATCH * TSTEPS * INW) {
    int  c = (int)(i & (INW - 1));
    long r = i >> 6;
    int  t = (int)(r & (TSTEPS - 1));
    long b = r >> 9;
    x_bf[((long)t * BATCH + b) * INW + c] = (bf16)x[i];
  }
}

__global__ void cast_kernel(const float* __restrict__ src, bf16* __restrict__ dst, long n) {
  long i = (long)blockIdx.x * blockDim.x + threadIdx.x;
  if (i < n) dst[i] = (bf16)src[i];
}

// out[b][o] = h1[b] . fcW[o] + fcb[o]    (grid 12, block 256)
__global__ void fc_kernel(const float* __restrict__ h, const float* __restrict__ fcW,
                          const float* __restrict__ fcb, float* __restrict__ out) {
  int o = blockIdx.x;
  int b = threadIdx.x;
  float s = fcb[o];
  const float* hr = h + (size_t)b * HID;
  const float* wr = fcW + (size_t)o * HID;
  for (int k = 0; k < HID; ++k) s += hr[k] * wr[k];
  out[(size_t)b * 12 + o] = s;
}

// ---------------------------------------------------------------------------
extern "C" void kernel_launch(void* const* d_in, const int* in_sizes, int n_in,
                              void* d_out, int out_size, void* d_ws, size_t ws_size,
                              hipStream_t stream) {
  const float* x   = (const float*)d_in[0];
  const float* Wx0 = (const float*)d_in[1];
  const float* Wh0 = (const float*)d_in[2];
  const float* b0  = (const float*)d_in[3];
  const float* Wx1 = (const float*)d_in[4];
  const float* Wh1 = (const float*)d_in[5];
  const float* b1  = (const float*)d_in[6];
  const float* fcW = (const float*)d_in[7];
  const float* fcb = (const float*)d_in[8];
  float* out = (float*)d_out;

  char* ws = (char*)d_ws;
  size_t off = 0;
  auto carve = [&](size_t bytes) -> char* {
    char* p = ws + off;
    off = (off + bytes + 255) & ~(size_t)255;
    return p;
  };
  unsigned* cnt = (unsigned*)carve(256);
  bf16* x_bf  = (bf16*)carve((size_t)TSTEPS * BATCH * INW * 2);
  bf16* wx0_b = (bf16*)carve((size_t)4 * HID * INW * 2);
  bf16* wh0_b = (bf16*)carve((size_t)4 * HID * HID * 2);
  bf16* wx1_b = (bf16*)carve((size_t)4 * HID * HID * 2);
  bf16* wh1_b = (bf16*)carve((size_t)4 * HID * HID * 2);
  bf16* h0_bf = (bf16*)carve((size_t)2 * BATCH * HID * 2);
  bf16* h1_bf = (bf16*)carve((size_t)2 * BATCH * HID * 2);
  float* h1_f = (float*)carve((size_t)BATCH * HID * 4);

  const long n_hinit = 2L * BATCH * HID;
  init_kernel<<<(unsigned)((n_hinit + 255) / 256), 256, 0, stream>>>(cnt, h0_bf, h1_bf, n_hinit);

  const long n_x = (long)BATCH * TSTEPS * INW;
  convert_x_kernel<<<(unsigned)((n_x + 255) / 256), 256, 0, stream>>>(x, x_bf);

  const long n_wx0 = 4L * HID * INW;
  const long n_wh  = 4L * HID * HID;
  cast_kernel<<<(unsigned)((n_wx0 + 255) / 256), 256, 0, stream>>>(Wx0, wx0_b, n_wx0);
  cast_kernel<<<(unsigned)((n_wh  + 255) / 256), 256, 0, stream>>>(Wh0, wh0_b, n_wh);
  cast_kernel<<<(unsigned)((n_wh  + 255) / 256), 256, 0, stream>>>(Wx1, wx1_b, n_wh);
  cast_kernel<<<(unsigned)((n_wh  + 255) / 256), 256, 0, stream>>>(Wh1, wh1_b, n_wh);

  lstm_persistent<<<NWG_TOTAL, 256, 0, stream>>>(
      x_bf, wx0_b, wh0_b, wx1_b, wh1_b, b0, b1, h0_bf, h1_bf, h1_f, cnt, 0);

  fc_kernel<<<12, 256, 0, stream>>>(h1_f, fcW, fcb, out);
}
